// TimescaleRNN_18236431139464
// MI455X (gfx1250) — matrix operand order repaired and checked by reading, not compiled, not run on hardware
//
#include <hip/hip_runtime.h>
#include <hip/hip_bf16.h>
#include <math.h>

// Problem constants (from reference): SEQ=512, B=64, D_IN=256, H=1024
#define SQ   512
#define BB   64
#define DD   256
#define HH   1024

#define NWG  16      // persistent workgroups for the scan
#define TPB  256     // 8 waves of 32
#define COLS 64      // H-columns owned per workgroup (16*64 = 1024)

typedef __attribute__((ext_vector_type(16))) __bf16 v16bf;
typedef __attribute__((ext_vector_type(8)))  __bf16 v8bf;
typedef __attribute__((ext_vector_type(8)))  float  v8f;

// ---------------------------------------------------------------------------
// Grid-wide barrier: monotonically increasing counter (reset by memsetAsync
// each launch -> deterministic across graph replays).
// ---------------------------------------------------------------------------
__device__ __forceinline__ void grid_sync(unsigned* cnt, unsigned target) {
    __syncthreads();
    if (threadIdx.x == 0) {
        __threadfence();                                   // flush h writes dev-wide
        __atomic_fetch_add(cnt, 1u, __ATOMIC_ACQ_REL);
        while (__atomic_load_n(cnt, __ATOMIC_ACQUIRE) < target) {
            __builtin_amdgcn_s_sleep(1);
        }
    }
    __syncthreads();
}

// ---------------------------------------------------------------------------
// Kernel 1: x_proj = x @ W_in^T + bias   ([32768 x 256] * [256 x 1024])
// One wave computes one 16x16 M-tile against two adjacent 16-wide N-tiles.
// bf16 WMMA, fp32 accumulate; result written fp32 into d_out (reused as the
// per-timestep input slab for the scan kernel).
// ---------------------------------------------------------------------------
__global__ void __launch_bounds__(TPB) xproj_kernel(
    const float* __restrict__ x,     // [S*B][D]
    const float* __restrict__ Win,   // [H][D]
    const float* __restrict__ bias,  // [H]
    float* __restrict__ out)         // [S*B][H]
{
    const int tid  = threadIdx.x;
    const int wave = tid >> 5;
    const int lane = tid & 31;
    const int half = lane >> 4;      // 0: lanes 0-15, 1: lanes 16-31
    const int l16  = lane & 15;

    const int gw    = blockIdx.x * 8 + wave;   // 0 .. 65535
    const int mtile = gw >> 5;                 // 0 .. 2047  (M = S*B / 16)
    const int ng    = gw & 31;                 // 0 .. 31 -> two N tiles each
    const int n0    = ng * 32 + l16;
    const int n1    = n0 + 16;
    const int arow  = mtile * 16 + l16;        // A row for this lane

    v8f c0 = {}; v8f c1 = {};
    const float* ab  = x   + (size_t)arow * DD;
    const float* b0r = Win + (size_t)n0   * DD;
    const float* b1r = Win + (size_t)n1   * DD;

#pragma unroll
    for (int kb = 0; kb < DD; kb += 32) {
        // A fragment: 16 contiguous K at kb + half*16
        v16bf a;
        const float* p0 = ab + kb + half * 16;
        const float* p1 = ab + kb + half * 16 + 8;
#pragma unroll
        for (int i = 0; i < 8; ++i) { a[i] = (__bf16)p0[i]; a[i + 8] = (__bf16)p1[i]; }
        // B fragment: lane = column n, 16 contiguous K at kb + half*16
        v16bf b0, b1;
        const float* q0 = b0r + kb + half * 16;
        const float* q1 = b1r + kb + half * 16;
#pragma unroll
        for (int i = 0; i < 16; ++i) { b0[i] = (__bf16)q0[i]; b1[i] = (__bf16)q1[i]; }

        c0 = __builtin_amdgcn_wmma_f32_16x16x32_bf16(false, a, false, b0,
                                                     (short)0, c0, false, false);
        c1 = __builtin_amdgcn_wmma_f32_16x16x32_bf16(false, a, false, b1,
                                                     (short)0, c1, false, false);
    }

    const float bi0 = bias[n0], bi1 = bias[n1];
#pragma unroll
    for (int r = 0; r < 8; ++r) {
        const int m = mtile * 16 + half * 8 + r;   // C layout: M = r + 8*half
        out[(size_t)m * HH + n0] = c0[r] + bi0;
        out[(size_t)m * HH + n1] = c1[r] + bi1;
    }
}

// ---------------------------------------------------------------------------
// Kernel 2: persistent recurrence.
//   h_t = (1-alpha)*h_{t-1} + alpha*tanh(xproj_t + h_{t-1} @ W_hh^T)
// 16 WGs; WG g owns columns [g*64, g*64+64). Its W_hh slice (64 contiguous
// rows of W_hh) is converted to bf16 and kept in 128 KB of LDS for all 512
// steps. h state ping-pongs between global bf16 buffers (GEMM operand) and
// fp32 buffers (elementwise blend); one grid barrier per step.
// ---------------------------------------------------------------------------
__global__ void __launch_bounds__(TPB, 1) rnn_scan_kernel(
    float*  __restrict__ out,     // [S][B][H], holds xproj on entry, h_t on exit
    float*  __restrict__ hn,      // [B][H] final state
    const float* __restrict__ Whh,   // [H][H]
    const float* __restrict__ alpha, // [H]
    float*  __restrict__ hf0, float* __restrict__ hf1,   // fp32 h ping-pong
    __bf16* __restrict__ hb0, __bf16* __restrict__ hb1,  // bf16 h ping-pong
    unsigned* __restrict__ sync_cnt)
{
    extern __shared__ __bf16 ldsw[];   // [COLS][HH] = 128 KB bf16 W_hh slice
    const int tid   = threadIdx.x;
    const int wg    = blockIdx.x;
    const int nbase = wg * COLS;

    // ---- preload W_hh slice (rows nbase..nbase+63 are contiguous fp32) ----
    {
        const float4* src = (const float4*)(Whh + (size_t)nbase * HH);
        const int nv = COLS * HH / 4;          // 16384 float4
        for (int i = tid; i < nv; i += TPB) {
            float4 v = src[i];
            __bf16* d = ldsw + (size_t)i * 4;
            d[0] = (__bf16)v.x; d[1] = (__bf16)v.y;
            d[2] = (__bf16)v.z; d[3] = (__bf16)v.w;
        }
    }
    // ---- zero h state (split across all WGs) ----
    {
        const __bf16 z = (__bf16)0.0f;
        for (int i = wg * TPB + tid; i < BB * HH; i += NWG * TPB) {
            hf0[i] = 0.0f; hf1[i] = 0.0f; hb0[i] = z; hb1[i] = z;
        }
    }
    unsigned epoch = NWG;
    grid_sync(sync_cnt, epoch);

    // ---- static wave -> tile mapping: 4 M-tiles x 4 N-tiles, 2 tiles/wave ----
    const int wave = tid >> 5, lane = tid & 31;
    const int half = lane >> 4, l16 = lane & 15;
    const int mt    = wave & 3;                 // M tile (B=64 -> 4 tiles)
    const int nloc0 = (wave >> 2) * 32 + l16;   // local column, tile 0
    const int nloc1 = nloc0 + 16;               // local column, tile 1
    const int n0    = nbase + nloc0;
    const int n1    = nbase + nloc1;
    const float al0 = alpha[n0], al1 = alpha[n1];
    const int arow  = mt * 16 + l16;

    for (int t = 0; t < SQ; ++t) {
        const __bf16* hread  = (t & 1) ? hb1 : hb0;
        __bf16*       hwrite = (t & 1) ? hb0 : hb1;
        const float*  hfr    = (t & 1) ? hf1 : hf0;
        float*        hfw    = (t & 1) ? hf0 : hf1;

        v8f c0 = {}; v8f c1 = {};
        const __bf16* abase = hread + (size_t)arow * HH;
#pragma unroll 4
        for (int kb = 0; kb < HH; kb += 32) {
            v16bf a;
            v8bf lo = *(const v8bf*)(abase + kb + half * 16);
            v8bf hi = *(const v8bf*)(abase + kb + half * 16 + 8);
#pragma unroll
            for (int i = 0; i < 8; ++i) { a[i] = lo[i]; a[i + 8] = hi[i]; }

            const __bf16* bp0 = ldsw + (size_t)nloc0 * HH + kb + half * 16;
            const __bf16* bp1 = ldsw + (size_t)nloc1 * HH + kb + half * 16;
            v8bf b0l = *(const v8bf*)(bp0);
            v8bf b0h = *(const v8bf*)(bp0 + 8);
            v8bf b1l = *(const v8bf*)(bp1);
            v8bf b1h = *(const v8bf*)(bp1 + 8);
            v16bf b0, b1;
#pragma unroll
            for (int i = 0; i < 8; ++i) {
                b0[i] = b0l[i]; b0[i + 8] = b0h[i];
                b1[i] = b1l[i]; b1[i + 8] = b1h[i];
            }
            c0 = __builtin_amdgcn_wmma_f32_16x16x32_bf16(false, a, false, b0,
                                                         (short)0, c0, false, false);
            c1 = __builtin_amdgcn_wmma_f32_16x16x32_bf16(false, a, false, b1,
                                                         (short)0, c1, false, false);
        }

        // ---- epilogue: h_new = (1-a)*h + a*tanh(xproj + y) ----
        const size_t tb = (size_t)t * BB * HH;
#pragma unroll
        for (int r = 0; r < 8; ++r) {
            const int m = mt * 16 + half * 8 + r;
            {
                const size_t oi = tb + (size_t)m * HH + n0;
                const int    hi = m * HH + n0;
                float ht = tanhf(out[oi] + c0[r]);
                float hv = (1.0f - al0) * hfr[hi] + al0 * ht;
                out[oi] = hv; hfw[hi] = hv; hwrite[hi] = (__bf16)hv;
                if (t == SQ - 1) hn[hi] = hv;
            }
            {
                const size_t oi = tb + (size_t)m * HH + n1;
                const int    hi = m * HH + n1;
                float ht = tanhf(out[oi] + c1[r]);
                float hv = (1.0f - al1) * hfr[hi] + al1 * ht;
                out[oi] = hv; hfw[hi] = hv; hwrite[hi] = (__bf16)hv;
                if (t == SQ - 1) hn[hi] = hv;
            }
        }

        epoch += NWG;
        grid_sync(sync_cnt, epoch);   // writes of h_t visible before step t+1
    }
}

// ---------------------------------------------------------------------------
// Launch
// ---------------------------------------------------------------------------
extern "C" void kernel_launch(void* const* d_in, const int* in_sizes, int n_in,
                              void* d_out, int out_size, void* d_ws, size_t ws_size,
                              hipStream_t stream) {
    const float* x     = (const float*)d_in[0];  // [S][B][D]
    const float* Win   = (const float*)d_in[1];  // [H][D]
    const float* Whh   = (const float*)d_in[2];  // [H][H]
    const float* bias  = (const float*)d_in[3];  // [H]
    const float* alpha = (const float*)d_in[4];  // [H]

    float* out = (float*)d_out;                       // [S][B][H]
    float* hn  = out + (size_t)SQ * BB * HH;          // [B][H]

    // workspace: sync counter + h ping-pong buffers (~768 KB)
    char*     ws       = (char*)d_ws;
    unsigned* sync_cnt = (unsigned*)ws;
    float*    hf0      = (float*)(ws + 256);
    float*    hf1      = hf0 + (size_t)BB * HH;
    __bf16*   hb0      = (__bf16*)(hf1 + (size_t)BB * HH);
    __bf16*   hb1      = hb0 + (size_t)BB * HH;

    hipMemsetAsync(sync_cnt, 0, 256, stream);

    // x_proj GEMM: 2048 M-tiles * 32 N-tile-pairs = 65536 waves = 8192 blocks
    xproj_kernel<<<8192, TPB, 0, stream>>>(x, Win, bias, out);

    // persistent scan: 16 WGs, 128 KB dynamic LDS each
    rnn_scan_kernel<<<NWG, TPB, COLS * HH * sizeof(__bf16), stream>>>(
        out, hn, Whh, alpha, hf0, hf1, hb0, hb1, sync_cnt);
}
